// FactorNet_3level220909_75703093559362
// MI455X (gfx1250) — compile-verified, matrix operand order checked by reading, not run on hardware
//
#include <hip/hip_runtime.h>
#include <cstdint>
#include <cstddef>

// ---------------------------------------------------------------------------
// FactorNet 3-level pipeline for MI455X (gfx1250, wave32, WMMA).
// GEMMs: v_wmma_f32_16x16x32_bf16, double-buffered LDS tiles, weight tiles
// staged with global_load_async_to_lds_b128 (ASYNCcnt), activations staged
// through VGPRs with fused BN/ReLU. BN stats fused into epilogues.
// ---------------------------------------------------------------------------

typedef __attribute__((ext_vector_type(8)))  float  v8f;
typedef __attribute__((ext_vector_type(16))) __bf16 v16bf;

union Frag { v16bf v; uint4 q[2]; };

__device__ __forceinline__ unsigned short f2bf(float f) {
  union { float f; unsigned int u; } v; v.f = f;
  unsigned int r = v.u + 0x7FFFu + ((v.u >> 16) & 1u);   // RNE
  return (unsigned short)(r >> 16);
}
__device__ __forceinline__ float bf2f(unsigned short s) {
  union { unsigned int u; float f; } v; v.u = ((unsigned int)s) << 16;
  return v.f;
}
__device__ __forceinline__ float sigm(float x) { return 1.0f / (1.0f + __expf(-x)); }

// ============================= constants ===================================
#define NB      16384          // batch
#define NC      16             // channels
#define NT      30             // time
#define NH      64             // hidden
#define CH1     320            // feature channels (2*120 + 5*16)
#define NPAIR   120
#define GEMM_BM 128
#define GEMM_BN 64
#define GEMM_BK 32
#define LPAD    48             // LDS row pitch in bf16 elems (96B, 16B aligned)

// ======================= init / small kernels ==============================

__global__ __launch_bounds__(256) void init_out_kernel(float* __restrict__ out,
                                                       const float* __restrict__ linb) {
  int i = blockIdx.x * 256 + threadIdx.x;
  if (i < NB) out[i] = linb[0];
}

__global__ __launch_bounds__(256) void branch_init_kernel(
    float* __restrict__ cstate, unsigned short* __restrict__ hcur,
    const float* __restrict__ cw, const float* __restrict__ wih,
    const float* __restrict__ whh, const float* __restrict__ bih,
    const float* __restrict__ bhh,
    unsigned short* __restrict__ W1, unsigned short* __restrict__ W2,
    unsigned short* __restrict__ W3, float* __restrict__ blstm,
    float* __restrict__ fstat, float* __restrict__ cstat, float* __restrict__ hstat) {
  int idx = blockIdx.x * 256 + threadIdx.x;          // grid covers NB*NH
  if (idx < NB * NH) { cstate[idx] = 0.0f; hcur[idx] = 0; }
  if (idx < CH1 * CH1)      W1[idx] = f2bf(cw[idx]);
  if (idx < 4 * NH * CH1)   W2[idx] = f2bf(wih[idx]);
  if (idx < 4 * NH * NH)    W3[idx] = f2bf(whh[idx]);
  if (idx < 4 * NH)         blstm[idx] = bih[idx] + bhh[idx];
  if (idx < 2 * CH1) { fstat[idx] = 0.0f; cstat[idx] = 0.0f; }
  if (idx < 2 * NH)         hstat[idx] = 0.0f;
}

__global__ void bn_finalize_kernel(const float* __restrict__ stat,
                                   const float* __restrict__ gamma,
                                   const float* __restrict__ beta,
                                   int n, float count,
                                   float* __restrict__ sc, float* __restrict__ sh) {
  int i = threadIdx.x;
  if (i >= n) return;
  float mean = stat[i] / count;
  float var  = stat[n + i] / count - mean * mean;    // population variance (jnp.var)
  float s = gamma[i] * rsqrtf(var + 1e-5f);
  sc[i] = s;
  sh[i] = beta[i] - mean * s;
}

// ======================= feature extraction ================================
// One block (128 threads) per batch row. x[b] (16x30) staged in LDS.
// Writes f as bf16 [b*Tp+t][320] and accumulates BN stats.

__global__ __launch_bounds__(128) void extract_kernel(
    const float* __restrict__ x, unsigned short* __restrict__ fA,
    float* __restrict__ fstat, int d, int Tp) {
  __shared__ float xs[NC * NT];
  __shared__ float sm[160], sstd[160];      // Tp*16 <= 160
  __shared__ float ssum[CH1], ssq[CH1];

  const int b = blockIdx.x, tid = threadIdx.x;
  for (int i = tid; i < NC * NT; i += 128) xs[i] = x[(size_t)b * NC * NT + i];
  for (int i = tid; i < CH1; i += 128) { ssum[i] = 0.0f; ssq[i] = 0.0f; }
  __syncthreads();

  const float invd  = 1.0f / (float)d;
  const float invd1 = 1.0f / (float)(d - 1);
  const float wnorm = 2.0f / (float)(d * (d + 1));

  // per (t, c): mean / std / ret / dec / zscore
  for (int task = tid; task < Tp * NC; task += 128) {
    int t = task >> 4, c = task & 15;
    const float* w = &xs[c * NT + t * d];
    float s = 0.0f;
    for (int j = 0; j < d; ++j) s += w[j];
    float m = s * invd;
    float ss = 0.0f;
    for (int j = 0; j < d; ++j) { float cc = w[j] - m; ss += cc * cc; }
    float sd  = sqrtf(ss * invd1);
    float ret = w[d - 1] / (w[0] + 1e-8f) - 1.0f;
    float dec = 0.0f;
    for (int j = 0; j < d; ++j) dec += w[j] * (float)(j + 1);
    dec *= wnorm;
    float z = m / (sd + 1e-8f);
    sm[task] = m; sstd[task] = sd;

    size_t base = ((size_t)b * Tp + t) * CH1;
    fA[base + 240 + c] = f2bf(sd);
    fA[base + 256 + c] = f2bf(z);
    fA[base + 272 + c] = f2bf(ret);
    fA[base + 288 + c] = f2bf(dec);
    fA[base + 304 + c] = f2bf(m);
    atomicAdd(&ssum[240 + c], sd);  atomicAdd(&ssq[240 + c], sd * sd);
    atomicAdd(&ssum[256 + c], z);   atomicAdd(&ssq[256 + c], z * z);
    atomicAdd(&ssum[272 + c], ret); atomicAdd(&ssq[272 + c], ret * ret);
    atomicAdd(&ssum[288 + c], dec); atomicAdd(&ssq[288 + c], dec * dec);
    atomicAdd(&ssum[304 + c], m);   atomicAdd(&ssq[304 + c], m * m);
  }
  __syncthreads();

  // per (t, pair): cov / corr
  for (int task = tid; task < Tp * NPAIR; task += 128) {
    int t = task / NPAIR, p = task % NPAIR;
    int i = 0, rem = p;
    while (rem >= NC - 1 - i) { rem -= NC - 1 - i; ++i; }
    int j = i + 1 + rem;
    float mi = sm[t * 16 + i], mj = sm[t * 16 + j];
    const float* wi = &xs[i * NT + t * d];
    const float* wj = &xs[j * NT + t * d];
    float s = 0.0f;
    for (int q = 0; q < d; ++q) s += (wi[q] - mi) * (wj[q] - mj);
    float cov  = s * invd1;
    float corr = cov / (sstd[t * 16 + i] * sstd[t * 16 + j] + 1e-8f);
    size_t base = ((size_t)b * Tp + t) * CH1;
    fA[base + p]         = f2bf(corr);
    fA[base + NPAIR + p] = f2bf(cov);
    atomicAdd(&ssum[p], corr);          atomicAdd(&ssq[p], corr * corr);
    atomicAdd(&ssum[NPAIR + p], cov);   atomicAdd(&ssq[NPAIR + p], cov * cov);
  }
  __syncthreads();
  for (int i = tid; i < CH1; i += 128) {
    atomicAdd(&fstat[i], ssum[i]);
    atomicAdd(&fstat[CH1 + i], ssq[i]);
  }
}

// ============================ WMMA GEMM ====================================
// C[M, N_total] = xform(A[M,K]) * W[N_total,K]^T  (bf16 in, fp32 accum)
// AMODE: 0 = raw, 1 = affine per-k, 2 = affine+relu per-k
// EPI:   0 = conv (+bias, stats, ->bf16), 1 = ->bf16, 2 = lstm gates (+xw+bias)
// Double-buffered LDS; B tiles via global_load_async_to_lds_b128 (ASYNCcnt);
// A tiles via VGPRs with fused per-k affine/ReLU.

__device__ __forceinline__ uint4 xform8(uint4 din, const float* sc, const float* sh,
                                        bool dorelu) {
  union { uint4 q; unsigned short s[8]; } u; u.q = din;
#pragma unroll
  for (int i = 0; i < 8; ++i) {
    float v = bf2f(u.s[i]) * sc[i] + sh[i];
    if (dorelu) v = v > 0.0f ? v : 0.0f;
    u.s[i] = f2bf(v);
  }
  return u.q;
}

__device__ __forceinline__ void async_b128_to_lds(const unsigned short* gptr,
                                                  unsigned lds_off) {
  // GLOBAL_LOAD_ASYNC_TO_LDS_B128 (GV mode): per-lane 64-bit addr, LDS dest.
  asm volatile("global_load_async_to_lds_b128 %0, %1, off"
               :: "v"(lds_off), "v"(gptr)
               : "memory");
}
__device__ __forceinline__ void wait_async0() {
  asm volatile("s_wait_asynccnt 0x0" ::: "memory");
}

template <int K, int AMODE, int EPI>
__global__ __launch_bounds__(256) void gemm_kernel(
    const unsigned short* __restrict__ A,
    const unsigned short* __restrict__ W,
    const float* __restrict__ ascale, const float* __restrict__ ashift,
    const float* __restrict__ bias,
    unsigned short* __restrict__ outbf,
    float* __restrict__ statbuf,
    const unsigned short* __restrict__ xwin, int xw_rowstride,
    float* __restrict__ gout,
    int N_total) {
  __shared__ __align__(16) unsigned short As[2][GEMM_BM * LPAD];
  __shared__ __align__(16) unsigned short Bs[2][GEMM_BN * LPAD];
  __shared__ float ssum[GEMM_BN], ssq[GEMM_BN];

  const int tid  = threadIdx.x;
  const int wave = tid >> 5, lane = tid & 31;
  const int bm = blockIdx.x * GEMM_BM;
  const int bn = blockIdx.y * GEMM_BN;

  if constexpr (EPI == 0) {
    for (int i = tid; i < GEMM_BN; i += 256) { ssum[i] = 0.0f; ssq[i] = 0.0f; }
  }

  // staging index maps
  const int ar = tid >> 1, ach = (tid & 1) * 16;     // A: 128 rows x 2 halves
  const int wn = tid >> 2, wkq = (tid & 3) * 8;      // B: 64 rows x 4 quarters
  const unsigned short* arow = A + (size_t)(bm + ar) * K + ach;
  const unsigned short* wrow = W + (size_t)(bn + wn) * K + wkq;

  auto issueB = [&](int kk, int buf) {
    async_b128_to_lds(wrow + kk,
                      (unsigned)(uintptr_t)&Bs[buf][wn * LPAD + wkq]);
  };
  auto loadA = [&](int kk, uint4& d0, uint4& d1) {
    d0 = *(const uint4*)(arow + kk);
    d1 = *(const uint4*)(arow + kk + 8);
    if constexpr (AMODE != 0) {
      const float* sc = ascale + kk + ach;
      const float* sh = ashift + kk + ach;
      d0 = xform8(d0, sc, sh, AMODE == 2);
      d1 = xform8(d1, sc + 8, sh + 8, AMODE == 2);
    }
  };
  auto storeA = [&](int buf, uint4 d0, uint4 d1) {
    *(uint4*)&As[buf][ar * LPAD + ach]     = d0;
    *(uint4*)&As[buf][ar * LPAD + ach + 8] = d1;
  };

  v8f acc[4];
#pragma unroll
  for (int i = 0; i < 4; ++i)
#pragma unroll
    for (int j = 0; j < 8; ++j) acc[i][j] = 0.0f;

  // ---- pipeline prologue: stage tile 0 into buffer 0 ----
  issueB(0, 0);
  {
    uint4 a0, a1;
    loadA(0, a0, a1);
    storeA(0, a0, a1);
  }
  wait_async0();
  __syncthreads();

  constexpr int NITER = K / GEMM_BK;
  const int mrow  = wave * 16 + (lane & 15);
  const int khalf = (lane >> 4) * 8;

#pragma unroll
  for (int it = 0; it < NITER; ++it) {
    const int cur = it & 1, nxt = cur ^ 1;

    uint4 na0, na1;
    if (it + 1 < NITER) {               // prefetch next tile while computing
      issueB((it + 1) * GEMM_BK, nxt);
      loadA((it + 1) * GEMM_BK, na0, na1);
    }

    // fragments: preload A + all 4 B fragments, then 4 back-to-back WMMAs
    Frag a;
    a.q[0] = *(const uint4*)&As[cur][mrow * LPAD + khalf];
    a.q[1] = *(const uint4*)&As[cur][mrow * LPAD + 16 + khalf];
    Frag bf[4];
#pragma unroll
    for (int nt = 0; nt < 4; ++nt) {
      const int ncol = nt * 16 + (lane & 15);
      bf[nt].q[0] = *(const uint4*)&Bs[cur][ncol * LPAD + khalf];
      bf[nt].q[1] = *(const uint4*)&Bs[cur][ncol * LPAD + 16 + khalf];
    }
#pragma unroll
    for (int nt = 0; nt < 4; ++nt) {
      acc[nt] = __builtin_amdgcn_wmma_f32_16x16x32_bf16(
          false, a.v, false, bf[nt].v, (short)0, acc[nt], false, false);
    }

    if (it + 1 < NITER) storeA(nxt, na0, na1);
    wait_async0();
    __syncthreads();
  }

  // Epilogue. C layout: lanes 0-15 -> N=lane, M rows 0..7; lanes 16-31 -> M 8..15.
  const unsigned mbase = (unsigned)(bm + wave * 16 + ((lane < 16) ? 0 : 8));
  const int nlane = lane & 15;
#pragma unroll
  for (int nt = 0; nt < 4; ++nt) {
    const unsigned gn = (unsigned)(bn + nt * 16 + nlane);
    float ls = 0.0f, lq = 0.0f;
#pragma unroll
    for (int v = 0; v < 8; ++v) {
      const unsigned gm = mbase + v;
      float val = acc[nt][v];
      if constexpr (EPI == 0) {
        val += bias[gn];
        outbf[gm * (unsigned)N_total + gn] = f2bf(val);
        ls += val; lq += val * val;
      } else if constexpr (EPI == 1) {
        outbf[gm * (unsigned)N_total + gn] = f2bf(val);
      } else {
        val += bf2f(xwin[gm * (unsigned)xw_rowstride + gn]) + bias[gn];
        gout[gm * (unsigned)N_total + gn] = val;
      }
    }
    if constexpr (EPI == 0) {
      atomicAdd(&ssum[gn - bn], ls);
      atomicAdd(&ssq[gn - bn], lq);
    }
  }
  if constexpr (EPI == 0) {
    __syncthreads();
    for (int i = tid; i < GEMM_BN; i += 256) {
      atomicAdd(&statbuf[bn + i], ssum[i]);
      atomicAdd(&statbuf[N_total + bn + i], ssq[i]);
    }
  }
}

// ========================= LSTM pointwise cell =============================

__global__ __launch_bounds__(256) void lstm_pointwise_kernel(
    const float* __restrict__ g, float* __restrict__ cstate,
    unsigned short* __restrict__ hcur, unsigned short* __restrict__ h_all,
    float* __restrict__ hstat, int Tp, int t) {
  __shared__ float hsum[NH], hsq[NH];
  const int tid = threadIdx.x;
  if (tid < NH) { hsum[tid] = 0.0f; hsq[tid] = 0.0f; }
  __syncthreads();

  const unsigned idx = blockIdx.x * 256 + tid;   // b*64 + hid
  const unsigned b = idx >> 6, hid = idx & 63;
  const float gi = g[b * 256u + hid];
  const float gf = g[b * 256u + 64u + hid];
  const float gg = g[b * 256u + 128u + hid];
  const float go = g[b * 256u + 192u + hid];
  const float c  = cstate[idx];
  const float cn = sigm(gf) * c + sigm(gi) * tanhf(gg);
  const float h  = sigm(go) * tanhf(cn);
  cstate[idx] = cn;
  hcur[idx]   = f2bf(h);
  h_all[(b * (unsigned)Tp + t) * NH + hid] = f2bf(h);
  atomicAdd(&hsum[hid], h);
  atomicAdd(&hsq[hid], h * h);
  __syncthreads();
  if (tid < NH) {
    atomicAdd(&hstat[tid], hsum[tid]);
    atomicAdd(&hstat[NH + tid], hsq[tid]);
  }
}

// =========================== final linear ==================================

__global__ __launch_bounds__(64) void final_kernel(
    const unsigned short* __restrict__ h_all, const float* __restrict__ sc,
    const float* __restrict__ sh, const float* __restrict__ lw, int Tp,
    float* __restrict__ out) {
  __shared__ float red[NH];
  const unsigned b = blockIdx.x, hid = threadIdx.x;
  const float scale = sc[hid], shift = sh[hid];
  float s = 0.0f;
  for (int t = 0; t < Tp; ++t) {
    float h = bf2f(h_all[(b * (unsigned)Tp + t) * NH + hid]);
    s += (h * scale + shift) * lw[hid * Tp + t];
  }
  red[hid] = s;
  __syncthreads();
  for (int o = 32; o > 0; o >>= 1) {
    if (hid < o) red[hid] += red[hid + o];
    __syncthreads();
  }
  if (hid == 0) out[b] += red[0];
}

// ============================== launcher ===================================

extern "C" void kernel_launch(void* const* d_in, const int* in_sizes, int n_in,
                              void* d_out, int out_size, void* d_ws, size_t ws_size,
                              hipStream_t stream) {
  const float* x         = (const float*)d_in[0];
  const float* ts_gamma  = (const float*)d_in[1];
  const float* ts_beta   = (const float*)d_in[2];
  const float* conv_w    = (const float*)d_in[3];
  const float* conv_b    = (const float*)d_in[4];
  const float* cbn_gamma = (const float*)d_in[5];
  const float* cbn_beta  = (const float*)d_in[6];
  const float* w_ih      = (const float*)d_in[7];
  const float* w_hh      = (const float*)d_in[8];
  const float* b_ih      = (const float*)d_in[9];
  const float* b_hh      = (const float*)d_in[10];
  const float* bn2_gamma = (const float*)d_in[11];
  const float* bn2_beta  = (const float*)d_in[12];
  const float* lin_w     = (const float*)d_in[13];
  const float* lin_b     = (const float*)d_in[14];
  float* out = (float*)d_out;

  // ---- workspace layout (branches processed sequentially; buffers reused) ----
  char* ws = (char*)d_ws;
  size_t off = 0;
  auto alloc = [&](size_t bytes) -> void* {
    void* p = ws + off;
    off = (off + bytes + 255) & ~(size_t)255;
    return p;
  };
  const size_t MMAX = (size_t)NB * 10;                       // 163840 rows
  unsigned short* fA    = (unsigned short*)alloc(MMAX * CH1 * 2);   // features (bf16)
  unsigned short* Ybuf  = (unsigned short*)alloc(MMAX * CH1 * 2);   // conv out (bf16)
  unsigned short* xw    = fA;                                // reuse: fA dead after conv GEMM
  float*          gtmp  = (float*)alloc((size_t)NB * 256 * 4);      // gate pre-activations
  float*          cst   = (float*)alloc((size_t)NB * NH * 4);       // LSTM cell state
  unsigned short* hcur  = (unsigned short*)alloc((size_t)NB * NH * 2);
  unsigned short* h_all = (unsigned short*)alloc(MMAX * NH * 2);
  unsigned short* W1    = (unsigned short*)alloc((size_t)CH1 * CH1 * 2);
  unsigned short* W2    = (unsigned short*)alloc((size_t)4 * NH * CH1 * 2);
  unsigned short* W3    = (unsigned short*)alloc((size_t)4 * NH * NH * 2);
  float* blstm  = (float*)alloc(4 * NH * 4);
  float* fstat  = (float*)alloc(2 * CH1 * 4);
  float* cstat  = (float*)alloc(2 * CH1 * 4);
  float* hstat  = (float*)alloc(2 * NH * 4);
  float* fscale = (float*)alloc(CH1 * 4);
  float* fshift = (float*)alloc(CH1 * 4);
  float* cscale = (float*)alloc(CH1 * 4);
  float* cshift = (float*)alloc(CH1 * 4);
  float* h2sc   = (float*)alloc(NH * 4);
  float* h2sh   = (float*)alloc(NH * 4);
  (void)ws_size; (void)in_sizes; (void)n_in; (void)out_size;

  init_out_kernel<<<NB / 256, 256, 0, stream>>>(out, lin_b);

  const int days[3] = {10, 5, 3};
  const int offs[3] = {0, NH * 3, NH * (3 + 6)};   // lin_w column offsets per branch

  for (int k = 0; k < 3; ++k) {
    const int d = days[k], Tp = NT / d;
    const int M = NB * Tp;

    branch_init_kernel<<<NB * NH / 256, 256, 0, stream>>>(
        cst, hcur,
        conv_w + (size_t)k * CH1 * CH1, w_ih + (size_t)k * 4 * NH * CH1,
        w_hh + (size_t)k * 4 * NH * NH, b_ih + (size_t)k * 4 * NH,
        b_hh + (size_t)k * 4 * NH,
        W1, W2, W3, blstm, fstat, cstat, hstat);

    extract_kernel<<<NB, 128, 0, stream>>>(x, fA, fstat, d, Tp);

    bn_finalize_kernel<<<1, CH1, 0, stream>>>(
        fstat, ts_gamma + k * CH1, ts_beta + k * CH1, CH1, (float)M, fscale, fshift);

    // conv: Y = BN(f) @ conv_w^T + conv_b   (fused input-BN; output stats)
    gemm_kernel<CH1, 1, 0><<<dim3(M / GEMM_BM, CH1 / GEMM_BN), 256, 0, stream>>>(
        fA, W1, fscale, fshift, conv_b + k * CH1, Ybuf, cstat,
        nullptr, 0, nullptr, CH1);

    bn_finalize_kernel<<<1, CH1, 0, stream>>>(
        cstat, cbn_gamma + k * CH1, cbn_beta + k * CH1, CH1, (float)M, cscale, cshift);

    // lstm input: xw = relu(BN(Y)) @ w_ih^T
    gemm_kernel<CH1, 2, 1><<<dim3(M / GEMM_BM, 256 / GEMM_BN), 256, 0, stream>>>(
        Ybuf, W2, cscale, cshift, nullptr, xw, nullptr,
        nullptr, 0, nullptr, 256);

    for (int t = 0; t < Tp; ++t) {
      // gates = h @ w_hh^T + xw[:, t, :] + (b_ih + b_hh)
      gemm_kernel<NH, 0, 2><<<dim3(NB / GEMM_BM, 256 / GEMM_BN), 256, 0, stream>>>(
          hcur, W3, nullptr, nullptr, blstm, nullptr, nullptr,
          xw + (size_t)t * 256, Tp * 256, gtmp, 256);
      lstm_pointwise_kernel<<<NB * NH / 256, 256, 0, stream>>>(
          gtmp, cst, hcur, h_all, hstat, Tp, t);
    }

    bn_finalize_kernel<<<1, NH, 0, stream>>>(
        hstat, bn2_gamma + k * NH, bn2_beta + k * NH, NH, (float)M, h2sc, h2sh);

    final_kernel<<<NB, NH, 0, stream>>>(h_all, h2sc, h2sh, lin_w + offs[k], Tp, out);
  }
}
